// SelfAttention_23003844837560
// MI455X (gfx1250) — matrix operand order repaired and checked by reading, not compiled, not run on hardware
//
#include <hip/hip_runtime.h>
#include <hip/hip_bf16.h>

// ---------------------------------------------------------------------------
// Causal self-attention for B=4, S=4096, E=1024, A=64 on gfx1250 (CDNA5).
// f16 WMMA (v_wmma_f32_16x16x32_f16) with fp32 accumulation everywhere.
// Software-pipelined: operands for step i+1 are loaded while step i's WMMAs
// execute (only ~1 wave/SIMD at this problem size -> ILP must hide latency).
// ---------------------------------------------------------------------------

typedef __attribute__((ext_vector_type(4)))  _Float16 v4h;
typedef __attribute__((ext_vector_type(8)))  _Float16 v8h;
typedef __attribute__((ext_vector_type(16))) _Float16 v16h;
typedef __attribute__((ext_vector_type(8)))  float    v8f;

#define BB 4
#define SS 4096
#define EE 1024
#define AA 64

static __device__ __forceinline__ v16h cat8(v8h lo, v8h hi) {
  return __builtin_shufflevector(lo, hi, 0,1,2,3,4,5,6,7,8,9,10,11,12,13,14,15);
}
static __device__ __forceinline__ v8h ldg8(const _Float16* p) {
  return *reinterpret_cast<const v8h*>(p);
}
static __device__ __forceinline__ v8f wmma_f16(v16h a, v16h b, v8f c) {
  // D = A(16x32 f16) * B(32x16 f16) + C(16x16 f32)
  return __builtin_amdgcn_wmma_f32_16x16x32_f16(false, a, false, b, (short)0, c,
                                                false, false);
}
static __device__ __forceinline__ int imin(int a, int b) { return a < b ? a : b; }

// ---------------------------------------------------------------------------
// Kernel 0: fp32 -> f16 conversion (vectorized x4; all sizes divisible by 4)
// ---------------------------------------------------------------------------
__global__ __launch_bounds__(256) void cvt_f32_f16(const float* __restrict__ in,
                                                   _Float16* __restrict__ out,
                                                   int n4) {
  int i = blockIdx.x * blockDim.x + threadIdx.x;
  const int stride = gridDim.x * blockDim.x;
  const float4* in4 = reinterpret_cast<const float4*>(in);
  v4h* out4 = reinterpret_cast<v4h*>(out);
  for (; i < n4; i += stride) {
    float4 f = in4[i];
    v4h h;
    h[0] = (_Float16)f.x; h[1] = (_Float16)f.y;
    h[2] = (_Float16)f.z; h[3] = (_Float16)f.w;
    out4[i] = h;
  }
}

// ---------------------------------------------------------------------------
// Kernel 1: fused QKV projection.  q = X*Wq^T + bq  (likewise k, v).
// One wave computes 16 rows x 64 cols for q, k, v.  Block = 8 waves = 128 rows.
// Operands are loaded straight from global f16 in contiguous k order (k = 16*hi + s for A and B)
// (B col n == W row n, contiguous).  1-deep software pipeline over k-steps.
// v is written transposed: vT[b][a][s]  (so PV matmul B-operand is contiguous)
// ---------------------------------------------------------------------------
__global__ __launch_bounds__(256) void qkv_proj(
    const _Float16* __restrict__ Xh,   // [B*S, E]
    const _Float16* __restrict__ Wqh,  // [A, E]
    const _Float16* __restrict__ Wkh,  // [A, E]
    const _Float16* __restrict__ Wvh,  // [A, E]
    const float* __restrict__ bq, const float* __restrict__ bk,
    const float* __restrict__ bv,
    _Float16* __restrict__ Qh,   // [B*S, A]
    _Float16* __restrict__ Kh,   // [B*S, A]
    _Float16* __restrict__ VTh)  // [B, A, S]
{
  const int lane = threadIdx.x & 31;
  const int wave = threadIdx.x >> 5;           // 0..7
  const int hi   = lane >> 4;                  // lane-half select
  const int ln   = lane & 15;
  const int rowBase = blockIdx.x * 128 + wave * 16;  // flattened (b,s) row

  v8f accQ[4] = {}, accK[4] = {}, accV[4] = {};

  const _Float16* xrow = Xh + (size_t)(rowBase + ln) * EE;

  auto loadA = [&](int kk) -> v16h {
    const _Float16* pa = xrow + kk + hi * 16;
    return cat8(ldg8(pa), ldg8(pa + 8));
  };
  auto loadW = [&](const _Float16* W, int t, int kk) -> v16h {
    const _Float16* w = W + (size_t)(t * 16 + ln) * EE + kk + hi * 16;
    return cat8(ldg8(w), ldg8(w + 8));
  };

  // pipeline prologue: operands for kk = 0
  v16h a0 = loadA(0);
  v16h bq0[4], bk0[4], bv0[4];
#pragma unroll
  for (int t = 0; t < 4; ++t) bq0[t] = loadW(Wqh, t, 0);
#pragma unroll
  for (int t = 0; t < 4; ++t) bk0[t] = loadW(Wkh, t, 0);
#pragma unroll
  for (int t = 0; t < 4; ++t) bv0[t] = loadW(Wvh, t, 0);

#pragma unroll 2
  for (int kk = 0; kk < EE; kk += 32) {
    const int kn = (kk + 32 < EE) ? kk + 32 : kk;  // last iter: harmless reload
    // WGP-scope prefetch ~4 iterations ahead (X streams from L2 -> warm L0)
    __builtin_prefetch(xrow + kn + 128, 0, 3);     // global_prefetch_b8

    // issue next step's loads first so they overlap this step's WMMAs
    v16h a1 = loadA(kn);
    v16h bq1[4], bk1[4], bv1[4];
#pragma unroll
    for (int t = 0; t < 4; ++t) bq1[t] = loadW(Wqh, t, kn);
#pragma unroll
    for (int t = 0; t < 4; ++t) bk1[t] = loadW(Wkh, t, kn);
#pragma unroll
    for (int t = 0; t < 4; ++t) bv1[t] = loadW(Wvh, t, kn);

#pragma unroll
    for (int t = 0; t < 4; ++t) accQ[t] = wmma_f16(a0, bq0[t], accQ[t]);
#pragma unroll
    for (int t = 0; t < 4; ++t) accK[t] = wmma_f16(a0, bk0[t], accK[t]);
#pragma unroll
    for (int t = 0; t < 4; ++t) accV[t] = wmma_f16(a0, bv0[t], accV[t]);

    a0 = a1;
#pragma unroll
    for (int t = 0; t < 4; ++t) {
      bq0[t] = bq1[t]; bk0[t] = bk1[t]; bv0[t] = bv1[t];
    }
  }

  // Epilogue.  C layout: lane holds col n=ln, rows m = g + 8*hi.
  const int brow = rowBase / SS;
  const int srow = rowBase % SS;
#pragma unroll
  for (int t = 0; t < 4; ++t) {
    const int col = t * 16 + ln;
    const float vbq = bq[col], vbk = bk[col], vbv = bv[col];
#pragma unroll
    for (int g = 0; g < 8; ++g) {
      const int m = g + 8 * hi;
      const size_t idx = (size_t)(rowBase + m) * AA + col;
      Qh[idx] = (_Float16)(accQ[t][g] + vbq);
      Kh[idx] = (_Float16)(accK[t][g] + vbk);
    }
    // v transposed: lane writes 8 contiguous halves along s -> one b128 store
    v8h vv;
#pragma unroll
    for (int g = 0; g < 8; ++g) vv[g] = (_Float16)(accV[t][g] + vbv);
    _Float16* vt = VTh + ((size_t)(brow * AA + col)) * SS + (srow + 8 * hi);
    *reinterpret_cast<v8h*>(vt) = vv;
  }
}

// ---------------------------------------------------------------------------
// Kernel 2: causal flash attention.  One wave owns 16 query rows.
// Per 32-key step: 4 score WMMAs (Q.K^T) + online softmax + 4 PV WMMAs.
// Double-buffered K operands (no rotate movs): step(kbA->kbB), step(kbB->kbA).
// V operands are issued right after the score WMMAs so the softmax VALU work
// hides their latency.  P is re-laid C-layout -> A-layout via wave-private LDS.
// ---------------------------------------------------------------------------
__global__ __launch_bounds__(256) void flash_attn(
    const _Float16* __restrict__ Qh,   // [B*S, A]
    const _Float16* __restrict__ Kh,   // [B*S, A]
    const _Float16* __restrict__ VTh,  // [B, A, S]
    float* __restrict__ Out)           // [B, S, A] fp32
{
  __shared__ __align__(16) _Float16 pst[8][16][40];  // per-wave P staging

  const int lane = threadIdx.x & 31;
  const int wave = threadIdx.x >> 5;
  const int hi   = lane >> 4;
  const int ln   = lane & 15;
  const int rowBase = blockIdx.x * 128 + wave * 16;  // flattened (b,s)
  const int b   = rowBase / SS;
  const int qlo = rowBase % SS;                      // first query row in batch

  const _Float16* Kb = Kh + (size_t)b * SS * AA;
  const _Float16* Vb = VTh + (size_t)b * AA * SS;

  // Q A-operands for contraction chunks a=[0,32) and a=[32,64)
  const _Float16* qrow = Qh + (size_t)(rowBase + ln) * AA;
  v16h qa0, qa1;
  { const _Float16* p = qrow + 0  + hi * 16; qa0 = cat8(ldg8(p), ldg8(p + 8)); }
  { const _Float16* p = qrow + 32 + hi * 16; qa1 = cat8(ldg8(p), ldg8(p + 8)); }

  // K operand loader: two 16x16 key tiles, contraction chunks a=[0,32),[32,64)
  auto loadKops = [&](int n0, v16h (&kb)[4]) {
    const _Float16* kr0 = Kb + (size_t)imin(n0 + ln, SS - 1) * AA;
    const _Float16* kr1 = Kb + (size_t)imin(n0 + 16 + ln, SS - 1) * AA;
    kb[0] = cat8(ldg8(kr0 + hi * 16),      ldg8(kr0 + hi * 16 + 8));
    kb[1] = cat8(ldg8(kr0 + 32 + hi * 16), ldg8(kr0 + 32 + hi * 16 + 8));
    kb[2] = cat8(ldg8(kr1 + hi * 16),      ldg8(kr1 + hi * 16 + 8));
    kb[3] = cat8(ldg8(kr1 + 32 + hi * 16), ldg8(kr1 + 32 + hi * 16 + 8));
  };

  v8f O[4] = {};
  float mrow[8], lrow[8];
#pragma unroll
  for (int g = 0; g < 8; ++g) { mrow[g] = -1.0e30f; lrow[g] = 0.0f; }

  const float sc = 0.125f * 1.44269504088896340736f;  // log2(e)/sqrt(64)
  const int kmax = qlo + 15;                          // last key needed

  // One 32-key step: consume kbCur, prefetch kbNext for n0+32.
  auto step = [&](int n0, v16h (&kbCur)[4], v16h (&kbNext)[4]) {
    // ---- scores: two 16x16 tiles covering keys [n0, n0+32) ----
    v8f s0 = {}, s1 = {};
    s0 = wmma_f16(qa0, kbCur[0], s0);
    s0 = wmma_f16(qa1, kbCur[1], s0);
    s1 = wmma_f16(qa0, kbCur[2], s1);
    s1 = wmma_f16(qa1, kbCur[3], s1);

    // V operands for this step (independent of softmax) -- issue now
    v16h vb[4];
    const int vcol = imin(n0 + hi * 16, SS - 16);
#pragma unroll
    for (int t = 0; t < 4; ++t) {
      const _Float16* p = Vb + (size_t)(t * 16 + ln) * SS + vcol;
      vb[t] = cat8(ldg8(p), ldg8(p + 8));
    }
    // prefetch next step's K operands (row indices are clamped -> safe)
    loadKops(n0 + 32, kbNext);

    // scale into log2 domain
#pragma unroll
    for (int g = 0; g < 8; ++g) { s0[g] *= sc; s1[g] *= sc; }

    // causal mask (wave-uniform branch; EXEC stays all-ones)
    if (n0 + 31 > qlo) {
      const int d0 = (n0 + ln) - (qlo + 8 * hi);   // mask if key - qrow > 0
      const int d1 = d0 + 16;
#pragma unroll
      for (int g = 0; g < 8; ++g) {
        if (d0 > g) s0[g] = -1.0e30f;
        if (d1 > g) s1[g] = -1.0e30f;
      }
    }

    // row max across the 16 lanes of each half (lane owns 8 rows)
    float tm[8];
#pragma unroll
    for (int g = 0; g < 8; ++g) tm[g] = fmaxf(s0[g], s1[g]);
#pragma unroll
    for (int off = 1; off < 16; off <<= 1) {
#pragma unroll
      for (int g = 0; g < 8; ++g)
        tm[g] = fmaxf(tm[g], __shfl_xor(tm[g], off, 32));
    }

    float al[8];
#pragma unroll
    for (int g = 0; g < 8; ++g) {
      const float mnew = fmaxf(mrow[g], tm[g]);
      al[g] = exp2f(mrow[g] - mnew);
      mrow[g] = mnew;
      s0[g] = exp2f(s0[g] - mnew);
      s1[g] = exp2f(s1[g] - mnew);
    }

    // row sum of probabilities
    float ts[8];
#pragma unroll
    for (int g = 0; g < 8; ++g) ts[g] = s0[g] + s1[g];
#pragma unroll
    for (int off = 1; off < 16; off <<= 1) {
#pragma unroll
      for (int g = 0; g < 8; ++g) ts[g] += __shfl_xor(ts[g], off, 32);
    }
#pragma unroll
    for (int g = 0; g < 8; ++g) lrow[g] = lrow[g] * al[g] + ts[g];

    // rescale output accumulators
#pragma unroll
    for (int t = 0; t < 4; ++t)
#pragma unroll
      for (int g = 0; g < 8; ++g) O[t][g] *= al[g];

    // ---- P: C-layout f32 -> A-operand f16 via wave-private LDS ----
#pragma unroll
    for (int g = 0; g < 8; ++g) {
      pst[wave][g + 8 * hi][ln]      = (_Float16)s0[g];
      pst[wave][g + 8 * hi][ln + 16] = (_Float16)s1[g];
    }
    asm volatile("s_wait_dscnt 0" ::: "memory");
    const _Float16* pp = &pst[wave][ln][hi * 16];
    v16h aP = cat8(*reinterpret_cast<const v8h*>(pp),
                   *reinterpret_cast<const v8h*>(pp + 8));
    asm volatile("" ::: "memory");

    // ---- PV: O[t] += P(16x32) * V(32x16) per 16-wide a-tile ----
#pragma unroll
    for (int t = 0; t < 4; ++t) O[t] = wmma_f16(aP, vb[t], O[t]);
  };

  // double-buffered main loop (no operand rotate copies)
  v16h kbA[4], kbB[4];
  loadKops(0, kbA);  // pipeline prologue
  int n0 = 0;
  while (true) {
    step(n0, kbA, kbB);
    n0 += 32;
    if (n0 > kmax) break;
    step(n0, kbB, kbA);
    n0 += 32;
    if (n0 > kmax) break;
  }

  // epilogue: normalize and store fp32
  float rl[8];
#pragma unroll
  for (int g = 0; g < 8; ++g) rl[g] = 1.0f / lrow[g];
#pragma unroll
  for (int t = 0; t < 4; ++t) {
#pragma unroll
    for (int g = 0; g < 8; ++g) {
      const int m = g + 8 * hi;
      Out[(size_t)(rowBase + m) * AA + (t * 16 + ln)] = O[t][g] * rl[g];
    }
  }
}

// ---------------------------------------------------------------------------
// Host-side launcher
// ---------------------------------------------------------------------------
extern "C" void kernel_launch(void* const* d_in, const int* in_sizes, int n_in,
                              void* d_out, int out_size, void* d_ws, size_t ws_size,
                              hipStream_t stream) {
  const float* emb = (const float*)d_in[0];   // [B,S,E]
  const float* Wq  = (const float*)d_in[1];   // [A,E]
  const float* bq  = (const float*)d_in[2];   // [A]
  const float* Wk  = (const float*)d_in[3];
  const float* bk  = (const float*)d_in[4];
  const float* Wv  = (const float*)d_in[5];
  const float* bv  = (const float*)d_in[6];
  float* out = (float*)d_out;

  // workspace layout (f16 buffers), all offsets 16B-aligned
  char* ws = (char*)d_ws;
  size_t off = 0;
  _Float16* Xh  = (_Float16*)(ws + off); off += (size_t)BB * SS * EE * 2;  // 32 MB
  _Float16* Wqh = (_Float16*)(ws + off); off += (size_t)AA * EE * 2;
  _Float16* Wkh = (_Float16*)(ws + off); off += (size_t)AA * EE * 2;
  _Float16* Wvh = (_Float16*)(ws + off); off += (size_t)AA * EE * 2;
  _Float16* Qh  = (_Float16*)(ws + off); off += (size_t)BB * SS * AA * 2;
  _Float16* Kh  = (_Float16*)(ws + off); off += (size_t)BB * SS * AA * 2;
  _Float16* VTh = (_Float16*)(ws + off); off += (size_t)BB * SS * AA * 2;

  // 0) convert inputs to f16
  const int nEmb4 = BB * SS * EE / 4;
  const int nW4   = AA * EE / 4;
  cvt_f32_f16<<<2048, 256, 0, stream>>>(emb, Xh, nEmb4);
  cvt_f32_f16<<<64, 256, 0, stream>>>(Wq, Wqh, nW4);
  cvt_f32_f16<<<64, 256, 0, stream>>>(Wk, Wkh, nW4);
  cvt_f32_f16<<<64, 256, 0, stream>>>(Wv, Wvh, nW4);

  // 1) fused QKV projection: 128 rows / block
  qkv_proj<<<(BB * SS) / 128, 256, 0, stream>>>(Xh, Wqh, Wkh, Wvh, bq, bk, bv,
                                                Qh, Kh, VTh);

  // 2) causal flash attention: 128 query rows / block
  flash_attn<<<(BB * SS) / 128, 256, 0, stream>>>(Qh, Kh, VTh, out);
}
